// Model_47278999995048
// MI455X (gfx1250) — matrix operand order repaired and checked by reading, not compiled, not run on hardware
//
#include <hip/hip_runtime.h>
#include <hip/hip_bf16.h>
#include <math.h>

// ---------------------------------------------------------------------------
// Types for CDNA5 WMMA
// ---------------------------------------------------------------------------
typedef __attribute__((ext_vector_type(8)))  float  v8f;
typedef __attribute__((ext_vector_type(16))) __bf16 v16bf;
typedef __attribute__((ext_vector_type(2)))  __bf16 v2bf;
typedef __bf16 bf16;

// pack two fp32 -> two bf16 in one dword.
// Preference: native v_cvt_pk_bf16_f32 > v_perm_b32 round-to-nearest pack
// (3 VALU ops per pair) > bit-exact RNE software fallback.
#if __has_builtin(__builtin_amdgcn_cvt_pk_bf16_f32)
__device__ __forceinline__ unsigned packbf2(float a, float b) {
    v2bf r = __builtin_amdgcn_cvt_pk_bf16_f32(a, b);
    return __builtin_bit_cast(unsigned, r);
}
#elif __has_builtin(__builtin_amdgcn_perm)
__device__ __forceinline__ unsigned packbf2(float a, float b) {
    union { float f; unsigned u; } x, y; x.f = a; y.f = b;
    // round to nearest (ties away), then take the two high halves:
    // D[15:0] = (a+0x8000)[31:16] (src1 bytes 3,2), D[31:16] = (b+0x8000)[31:16]
    return __builtin_amdgcn_perm(y.u + 0x8000u, x.u + 0x8000u, 0x07060302u);
}
#else
__device__ __forceinline__ unsigned short bf_bits(float f) {
    union { float f; unsigned u; } x; x.f = f;
    unsigned u = x.u;
    unsigned r = u + 0x7fffu + ((u >> 16) & 1u);   // round-to-nearest-even
    return (unsigned short)(r >> 16);
}
__device__ __forceinline__ unsigned packbf2(float a, float b) {
    return (unsigned)bf_bits(a) | ((unsigned)bf_bits(b) << 16);
}
#endif

__device__ __forceinline__ float apply_act(float v, int act) {
    if (act == 1) return fmaxf(v, 0.f);
    if (act == 2) return 0.5f * v * (1.f + erff(v * 0.70710678118654752f));
    return v;
}

// ---------------------------------------------------------------------------
// Generic GEMM:  C = [Res +] act(A[M,K] @ W[K,N] + bias)
// fp32 in HBM -> bf16 in LDS -> v_wmma_f32_16x16x32_bf16.
// Block tile 64x64x64; 8 waves; each wave owns a 32x16 strip (2 acc tiles),
// 4 WMMAs per K-step per wave. Software-pipelined: global loads for tile k+1
// are in flight (registers) while tile k is multiplied out of LDS.
// REQUIREMENT: M % 64 == 0 and K % 64 == 0 (true for every GEMM here);
// N may be arbitrary (guarded slow path used only by the partial-N block).
// ---------------------------------------------------------------------------
#define BM 64
#define BN 64
#define BK 64

__global__ __launch_bounds__(256) void k_gemm_bf16(
    const float* __restrict__ A, const float* __restrict__ W,
    const float* __restrict__ bias, const float* __restrict__ Res,
    float* __restrict__ C, int M, int N, int K, int act,
    long long sA, long long sW, long long sB, long long sR, long long sC)
{
    const int batch = blockIdx.z;
    A += (long long)batch * sA;
    W += (long long)batch * sW;
    if (bias) bias += (long long)batch * sB;
    if (Res)  Res  += (long long)batch * sR;
    C += (long long)batch * sC;

    __shared__ bf16 As[BM * BK];     // row-major [m][k], row stride 64
    __shared__ bf16 Bs[BN * BK];     // N-major   [n][k], row stride 64

    const int tid  = threadIdx.x;
    const int lane = tid & 31;
    const int wave = tid >> 5;
    const int bm0 = blockIdx.y * BM;
    const int bn0 = blockIdx.x * BN;
    const int wm = (wave & 1) * 32;        // 0 or 32
    const int wn = (wave >> 1) * 16;       // 0,16,32,48

    const bool fullN = (bn0 + BN) <= N;    // uniform per block

    v8f acc0, acc1;
    #pragma unroll
    for (int i = 0; i < 8; i++) { acc0[i] = 0.f; acc1[i] = 0.f; }

    const int mrow = lane & 15;
    const int hi   = lane >> 4;

    // A staging: 4 float4 (rows lr, lr+16, lr+32, lr+48; 4 consecutive k each)
    float4 ar[4];
    const int lr  = tid >> 4;        // row   (0..15 base, +16 per i)
    const int lc4 = tid & 15;        // float4 column (0..15)
    // B staging: 2 k-pair groups, each = float4 along n from rows k and k+1
    float4 br0[2], br1[2];
    const int pkp = tid >> 4;        // k-pair index (0..15 base, +16 per i)
    const int pc4 = tid & 15;        // float4 column along n

    auto loadA = [&](int k0) {
        #pragma unroll
        for (int i = 0; i < 4; i++) {
            int r = lr + i * 16;
            ar[i] = *(const float4*)(A + (long long)(bm0 + r) * K + k0 + lc4 * 4);
        }
    };
    auto loadB_full = [&](int k0) {
        #pragma unroll
        for (int i = 0; i < 2; i++) {
            int kp = pkp + i * 16;
            const float* w0 = W + (long long)(k0 + 2 * kp) * N + bn0 + pc4 * 4;
            br0[i] = *(const float4*)w0;
            br1[i] = *(const float4*)(w0 + N);
        }
    };
    auto loadB_part = [&](int k0) {
        #pragma unroll
        for (int i = 0; i < 2; i++) {
            int kp = pkp + i * 16;
            const float* w0 = W + (long long)(k0 + 2 * kp) * N;
            const float* w1 = w0 + N;
            int gn = bn0 + pc4 * 4;
            float* p0 = &br0[i].x;
            float* p1 = &br1[i].x;
            #pragma unroll
            for (int j = 0; j < 4; j++) {
                p0[j] = (gn + j < N) ? w0[gn + j] : 0.f;
                p1[j] = (gn + j < N) ? w1[gn + j] : 0.f;
            }
        }
    };
    auto storeA = [&]() {
        #pragma unroll
        for (int i = 0; i < 4; i++) {
            int r = lr + i * 16;
            uint2 p; p.x = packbf2(ar[i].x, ar[i].y); p.y = packbf2(ar[i].z, ar[i].w);
            *(uint2*)((char*)As + r * (BK * 2) + lc4 * 8) = p;   // ds_store_b64
        }
    };
    auto storeB = [&]() {
        #pragma unroll
        for (int i = 0; i < 2; i++) {
            int kp = pkp + i * 16;
            int n0 = pc4 * 4;
            const float* p0 = &br0[i].x;
            const float* p1 = &br1[i].x;
            #pragma unroll
            for (int j = 0; j < 4; j++) {
                unsigned u = packbf2(p0[j], p1[j]);  // k even -> lo, k odd -> hi
                *(unsigned*)((char*)Bs + ((n0 + j) * BK + 2 * kp) * 2) = u;
            }
        }
    };
    auto compute = [&]() {
        #pragma unroll
        for (int h2 = 0; h2 < 2; h2++) {           // two 32-K halves
            const int kbB = h2 * 32 + hi * 16;
            v16bf bfrag;
            {
                const bf16* bp = &Bs[(wn + mrow) * BK + kbB];
                #pragma unroll
                for (int j = 0; j < 16; j++) bfrag[j] = bp[j];
            }
            const int kbA = h2 * 32 + hi * 16;
            v16bf afrag;
            {
                const bf16* ap = &As[(wm + mrow) * BK + kbA];
                #pragma unroll
                for (int j = 0; j < 8; j++) { afrag[j] = ap[j]; afrag[8 + j] = ap[8 + j]; }
            }
            acc0 = __builtin_amdgcn_wmma_f32_16x16x32_bf16(
                       false, afrag, false, bfrag, (short)0, acc0, false, false);
            {
                const bf16* ap = &As[(wm + 16 + mrow) * BK + kbA];
                #pragma unroll
                for (int j = 0; j < 8; j++) { afrag[j] = ap[j]; afrag[8 + j] = ap[8 + j]; }
            }
            acc1 = __builtin_amdgcn_wmma_f32_16x16x32_bf16(
                       false, afrag, false, bfrag, (short)0, acc1, false, false);
        }
    };

    // prologue: fill LDS with tile 0
    loadA(0);
    if (fullN) loadB_full(0); else loadB_part(0);
    storeA(); storeB();
    __syncthreads();

    for (int k0 = BK; k0 < K; k0 += BK) {
        // prefetch tile k0+BK into L2 (global_prefetch_b8)
        if (k0 + BK < K) {
            __builtin_prefetch(A + (long long)(bm0 + lr) * K + k0 + BK + lc4 * 4, 0, 0);
            __builtin_prefetch(W + (long long)(k0 + BK + lr) * N + bn0 + lc4 * 4, 0, 0);
        }
        // issue global loads for this tile (land after compute)
        loadA(k0);
        if (fullN) loadB_full(k0); else loadB_part(k0);
        compute();                 // multiplies previous tile out of LDS
        __syncthreads();
        storeA(); storeB();
        __syncthreads();
    }
    compute();                     // last tile

    // epilogue: lane stores one column, 8 rows per accumulator
    const int n = bn0 + wn + mrow;
    if (n < N) {
        float bv = bias ? bias[n] : 0.f;
        #pragma unroll
        for (int r = 0; r < 8; r++) {
            int m0 = bm0 + wm + r + 8 * hi;
            {
                float v = apply_act(acc0[r] + bv, act);
                if (Res) v += Res[(long long)m0 * N + n];
                C[(long long)m0 * N + n] = v;
            }
            int m1 = m0 + 16;
            {
                float v = apply_act(acc1[r] + bv, act);
                if (Res) v += Res[(long long)m1 * N + n];
                C[(long long)m1 * N + n] = v;
            }
        }
    }
}

// ---------------------------------------------------------------------------
// LayerNorm (one block per row)
// ---------------------------------------------------------------------------
__global__ __launch_bounds__(256) void k_layernorm(
    const float* __restrict__ in, float* __restrict__ out,
    const float* __restrict__ g, const float* __restrict__ b, int D)
{
    const int row = blockIdx.x;
    const float* x = in + (long long)row * D;
    float s = 0.f, s2 = 0.f;
    for (int d = threadIdx.x; d < D; d += 256) { float v = x[d]; s += v; s2 += v * v; }
    #pragma unroll
    for (int off = 16; off > 0; off >>= 1) { s += __shfl_xor(s, off); s2 += __shfl_xor(s2, off); }
    __shared__ float sh1[8], sh2[8];
    int wave = threadIdx.x >> 5, lane = threadIdx.x & 31;
    if (lane == 0) { sh1[wave] = s; sh2[wave] = s2; }
    __syncthreads();
    if (threadIdx.x == 0) {
        float a = 0.f, c = 0.f;
        for (int i = 0; i < 8; i++) { a += sh1[i]; c += sh2[i]; }
        sh1[0] = a; sh2[0] = c;
    }
    __syncthreads();
    float mean = sh1[0] / D;
    float var  = sh2[0] / D - mean * mean;
    float inv  = rsqrtf(var + 1e-5f);
    float* o = out + (long long)row * D;
    for (int d = threadIdx.x; d < D; d += 256)
        o[d] = (x[d] - mean) * inv * g[d] + b[d];
}

// ---------------------------------------------------------------------------
// Small-S multi-head attention: one block per (batch, head); S <= 12, hd = 128
// ---------------------------------------------------------------------------
__global__ __launch_bounds__(256) void k_attn(
    const float* __restrict__ qkv, float* __restrict__ out, int S, int D, int HD)
{
    const int b = blockIdx.x, h = blockIdx.y;
    __shared__ float qs[12 * 128], ks[12 * 128], vs[12 * 128], sc[12 * 12];
    const int tid = threadIdx.x;
    const int base = b * S;
    const int ld = 3 * D;
    for (int idx = tid; idx < S * HD; idx += 256) {
        int s = idx / HD, d = idx % HD;
        const float* row = qkv + (long long)(base + s) * ld;
        qs[idx] = row[h * HD + d];
        ks[idx] = row[D + h * HD + d];
        vs[idx] = row[2 * D + h * HD + d];
    }
    __syncthreads();
    float scale = rsqrtf((float)HD);
    for (int p = tid; p < S * S; p += 256) {
        int i = p / S, j = p % S;
        float acc = 0.f;
        for (int d = 0; d < HD; d++) acc += qs[i * HD + d] * ks[j * HD + d];
        sc[p] = acc * scale;
    }
    __syncthreads();
    if (tid < S) {
        int i = tid;
        float mx = -1e30f;
        for (int j = 0; j < S; j++) mx = fmaxf(mx, sc[i * S + j]);
        float sum = 0.f;
        for (int j = 0; j < S; j++) { float e = __expf(sc[i * S + j] - mx); sc[i * S + j] = e; sum += e; }
        float inv = 1.f / sum;
        for (int j = 0; j < S; j++) sc[i * S + j] *= inv;
    }
    __syncthreads();
    for (int idx = tid; idx < S * HD; idx += 256) {
        int i = idx / HD, d = idx % HD;
        float acc = 0.f;
        for (int j = 0; j < S; j++) acc += sc[i * S + j] * vs[j * HD + d];
        out[(long long)(base + i) * D + h * HD + d] = acc;
    }
}

// ---------------------------------------------------------------------------
// MoE: router (wave per token), ordered capacity scan, dispatch, combine
// ---------------------------------------------------------------------------
__global__ __launch_bounds__(256) void k_router(
    const float* __restrict__ X, const float* __restrict__ Wr, int N, int K,
    int* __restrict__ e1, int* __restrict__ e2,
    float* __restrict__ g1, float* __restrict__ g2)
{
    const int wave = threadIdx.x >> 5, lane = threadIdx.x & 31;
    const int tok = blockIdx.x * 8 + wave;
    if (tok >= N) return;
    const float* xr = X + (long long)tok * K;
    float a0 = 0.f, a1 = 0.f, a2 = 0.f, a3 = 0.f;
    for (int k = lane; k < K; k += 32) {
        float v = xr[k];
        const float* w = Wr + k * 4;
        a0 += v * w[0]; a1 += v * w[1]; a2 += v * w[2]; a3 += v * w[3];
    }
    #pragma unroll
    for (int off = 16; off > 0; off >>= 1) {
        a0 += __shfl_xor(a0, off); a1 += __shfl_xor(a1, off);
        a2 += __shfl_xor(a2, off); a3 += __shfl_xor(a3, off);
    }
    if (lane == 0) {
        float l[4] = {a0, a1, a2, a3};
        float mx = fmaxf(fmaxf(l[0], l[1]), fmaxf(l[2], l[3]));
        float p[4]; float sum = 0.f;
        for (int e = 0; e < 4; e++) { p[e] = __expf(l[e] - mx); sum += p[e]; }
        for (int e = 0; e < 4; e++) p[e] /= sum;
        int i1 = 0; float b1v = p[0];
        for (int e = 1; e < 4; e++) if (p[e] > b1v) { b1v = p[e]; i1 = e; }
        int i2 = 0; float b2v = -1.f;
        for (int e = 0; e < 4; e++) if (e != i1 && p[e] > b2v) { b2v = p[e]; i2 = e; }
        e1[tok] = i1; e2[tok] = i2; g1[tok] = b1v; g2[tok] = b2v;
    }
}

__global__ void k_moe_scan(
    const int* __restrict__ e1, const int* __restrict__ e2,
    const float* __restrict__ g1, const float* __restrict__ g2,
    int N, int cap,
    int* __restrict__ d1, int* __restrict__ d2,
    float* __restrict__ w1, float* __restrict__ w2)
{
    if (threadIdx.x != 0 || blockIdx.x != 0) return;
    int cnt1[4] = {0, 0, 0, 0};
    for (int n = 0; n < N; n++) {
        int e = e1[n];
        int p = cnt1[e]++;
        d1[n] = (p < cap) ? e * cap + p : -1;
        w1[n] = (p < cap) ? g1[n] : 0.f;
    }
    int cnt2[4] = {0, 0, 0, 0};
    for (int n = 0; n < N; n++) {
        if (g2[n] > 0.2f) {                      // threshold_eval
            int e = e2[n];
            int p = cnt2[e]++ + cnt1[e];         // stacked after top-1 counts
            d2[n] = (p < cap) ? e * cap + p : -1;
            w2[n] = (p < cap) ? g2[n] : 0.f;
        } else { d2[n] = -1; w2[n] = 0.f; }
    }
}

__global__ __launch_bounds__(256) void k_moe_dispatch(
    const float* __restrict__ X, const int* __restrict__ d1,
    const int* __restrict__ d2, float* __restrict__ EB, int D)
{
    const int n = blockIdx.x;
    const int s1 = d1[n], s2 = d2[n];
    const float* src = X + (long long)n * D;
    for (int d = threadIdx.x; d < D; d += 256) {
        float v = src[d];
        if (s1 >= 0) EB[(long long)s1 * D + d] = v;
        if (s2 >= 0) EB[(long long)s2 * D + d] = v;
    }
}

__global__ __launch_bounds__(256) void k_moe_combine(
    const float* __restrict__ X, const float* __restrict__ EO,
    const int* __restrict__ d1, const int* __restrict__ d2,
    const float* __restrict__ w1, const float* __restrict__ w2,
    float* __restrict__ out, int D)
{
    const int n = blockIdx.x;
    const int s1 = d1[n], s2 = d2[n];
    const float a = w1[n], b = w2[n];
    for (int d = threadIdx.x; d < D; d += 256) {
        float v = X[(long long)n * D + d];
        if (s1 >= 0) v += a * EO[(long long)s1 * D + d];
        if (s2 >= 0) v += b * EO[(long long)s2 * D + d];
        out[(long long)n * D + d] = v;
    }
}

// ---------------------------------------------------------------------------
// Elementwise helpers
// ---------------------------------------------------------------------------
__global__ __launch_bounds__(256) void k_img_input(
    const float* __restrict__ iv, const float* __restrict__ iemb,
    const float* __restrict__ femb, const int* __restrict__ iidx,
    const int* __restrict__ fidx, float* __restrict__ X, int B, int D)
{
    long long idx = (long long)blockIdx.x * 256 + threadIdx.x;
    if (idx >= (long long)B * D) return;
    int b = (int)(idx / D), d = (int)(idx % D);
    X[idx] = iv[idx] + iemb[(long long)iidx[b] * D + d] + femb[(long long)fidx[b] * D + d];
}

__global__ __launch_bounds__(256) void k_geo_embed(
    const int* __restrict__ geocode, const float* __restrict__ emb,
    float* __restrict__ X, int b0, int nb)
{
    long long idx = (long long)blockIdx.x * 256 + threadIdx.x;
    if (idx >= (long long)nb * 12 * 512) return;
    int d = (int)(idx % 512);
    int s = (int)((idx / 512) % 12);
    int bb = (int)(idx / (512 * 12));
    int code = geocode[(long long)(b0 + bb) * 12 + s];
    int i2 = d & ~1;
    float freq = __expf(-(float)i2 * (9.210340371976184f / 512.0f)); // ln(1e4)/512
    float ang = (float)s * freq;
    float pe = (d & 1) ? cosf(ang) : sinf(ang);
    X[idx] = emb[(long long)code * 512 + d] + pe;
}

__global__ __launch_bounds__(256) void k_meanpool(
    const float* __restrict__ X, float* __restrict__ out, int nb, int S, int D)
{
    long long idx = (long long)blockIdx.x * 256 + threadIdx.x;
    if (idx >= (long long)nb * D) return;
    int b = (int)(idx / D), d = (int)(idx % D);
    float s = 0.f;
    for (int j = 0; j < S; j++) s += X[((long long)b * S + j) * D + d];
    out[idx] = s * (1.0f / S);
}

__global__ __launch_bounds__(256) void k_build_merge(
    const float* __restrict__ vi, const float* __restrict__ vc,
    const float* __restrict__ vg, const float* __restrict__ me,
    float* __restrict__ Xm, int B, int D)
{
    long long idx = (long long)blockIdx.x * 256 + threadIdx.x;
    if (idx >= (long long)B * 3 * D) return;
    int d = (int)(idx % D);
    int j = (int)((idx / D) % 3);
    int b = (int)(idx / (3 * D));
    const float* src = (j == 0) ? vi : (j == 1) ? vc : vg;
    Xm[idx] = src[(long long)b * D + d] + me[j * D + d];
}

// ---------------------------------------------------------------------------
// Host-side parameter views & orchestration
// ---------------------------------------------------------------------------
struct HeadP { const float *b1, *b2, *lnb, *lng, *w1, *w2; };
struct EncP  { const float *bo, *bqkv, *fb1, *fb2, *fw1, *fw2,
                           *ln1b, *ln1g, *ln2b, *ln2g, *wo, *wqkv; };
struct SmbP  { const float *fa_b1, *fa_b2, *fa_w1, *fa_w2,
                           *fb_b1, *fb_b2, *fb_w1, *fb_w2,
                           *m_b1, *m_b2, *m_router, *m_w1, *m_w2; };
struct WS {
    float *X, *T1, *X2, *EB, *HB, *EO;
    float *VIMG, *VCONT, *VGEO, *GPOOL, *MPOOL;
    float *G1, *G2, *W1O, *W2O;
    int *E1, *E2, *D1, *D2;
};

static void gemm(hipStream_t st, const float* A, const float* W, const float* bias,
                 const float* Res, float* C, int M, int N, int K, int act,
                 int batches = 1, long long sA = 0, long long sW = 0,
                 long long sB = 0, long long sR = 0, long long sC = 0)
{
    dim3 grid((N + BN - 1) / BN, (M + BM - 1) / BM, batches);
    k_gemm_bf16<<<grid, 256, 0, st>>>(A, W, bias, Res, C, M, N, K, act, sA, sW, sB, sR, sC);
}

static void run_smb(hipStream_t st, const float* Xin, const SmbP& p, const WS& w)
{
    const int N = 2048, D = 1536, H = 4096, CAP = 1024;
    // ff_before + residual -> X2 (= x1)
    gemm(st, Xin, p.fb_w1, p.fb_b1, nullptr, w.T1, N, H, D, 2);
    gemm(st, w.T1, p.fb_w2, p.fb_b2, Xin, w.X2, N, D, H, 0);
    // routing
    k_router<<<N / 8, 256, 0, st>>>(w.X2, p.m_router, N, D, w.E1, w.E2, w.G1, w.G2);
    k_moe_scan<<<1, 32, 0, st>>>(w.E1, w.E2, w.G1, w.G2, N, CAP, w.D1, w.D2, w.W1O, w.W2O);
    k_moe_dispatch<<<N, 256, 0, st>>>(w.X2, w.D1, w.D2, w.EB, D);
    // experts (batched over grid.z)
    gemm(st, w.EB, p.m_w1, p.m_b1, nullptr, w.HB, CAP, H, D, 2, 4,
         (long long)CAP * D, (long long)D * H, H, 0, (long long)CAP * H);
    gemm(st, w.HB, p.m_w2, p.m_b2, nullptr, w.EO, CAP, D, H, 0, 4,
         (long long)CAP * H, (long long)H * D, D, 0, (long long)CAP * D);
    // combine + residual -> X (= x2)
    k_moe_combine<<<N, 256, 0, st>>>(w.X2, w.EO, w.D1, w.D2, w.W1O, w.W2O, w.X, D);
    // ff_after + residual -> X2 (= smb output)
    gemm(st, w.X, p.fa_w1, p.fa_b1, nullptr, w.T1, N, H, D, 2);
    gemm(st, w.T1, p.fa_w2, p.fa_b2, w.X, w.X2, N, D, H, 0);
}

static void run_head(hipStream_t st, const float* S, const HeadP& h, float* out,
                     int Din, int Dh, int Dout, int rows, float* Lbuf, float* Hbuf)
{
    k_layernorm<<<rows, 256, 0, st>>>(S, Lbuf, h.lng, h.lnb, Din);
    gemm(st, Lbuf, h.w1, h.b1, nullptr, Hbuf, rows, Dh, Din, 2);
    gemm(st, Hbuf, h.w2, h.b2, nullptr, out, rows, Dout, Dh, 0);
}

static void run_enc(hipStream_t st, float* X, float* BIG, float* AO, float* R,
                    const EncP& p, int T, int nb, int S)
{
    const int D = 512, HD = 128, DFF = 2048;
    gemm(st, X, p.wqkv, p.bqkv, nullptr, BIG, T, 3 * D, D, 0);           // qkv
    k_attn<<<dim3(nb, 4), 256, 0, st>>>(BIG, AO, S, D, HD);
    gemm(st, AO, p.wo, p.bo, X, R, T, D, D, 0);                          // x + mha(x)
    k_layernorm<<<T, 256, 0, st>>>(R, X, p.ln1g, p.ln1b, D);             // -> x1 (in X)
    gemm(st, X, p.fw1, p.fb1, nullptr, BIG, T, DFF, D, 1);               // relu ff
    gemm(st, BIG, p.fw2, p.fb2, X, R, T, D, DFF, 0);                     // x1 + ff(x1)
    k_layernorm<<<T, 256, 0, st>>>(R, X, p.ln2g, p.ln2b, D);             // -> next x
}

extern "C" void kernel_launch(void* const* d_in, const int* in_sizes, int n_in,
                              void* d_out, int out_size, void* d_ws, size_t ws_size,
                              hipStream_t stream)
{
    (void)in_sizes; (void)n_in; (void)out_size; (void)ws_size;
    const float* img_vec     = (const float*)d_in[0];
    const float* content_vec = (const float*)d_in[1];
    const float* query_vec   = (const float*)d_in[2];

    // params pytree, jax tree_flatten order (alphabetical keys, lists in order)
    int pi = 3;
    auto F = [&](int i) { return (const float*)d_in[i]; };
    auto getHead = [&]() { HeadP h; h.b1 = F(pi++); h.b2 = F(pi++); h.lnb = F(pi++);
                           h.lng = F(pi++); h.w1 = F(pi++); h.w2 = F(pi++); return h; };
    auto getEnc = [&]() { EncP e; e.bo = F(pi++); e.bqkv = F(pi++); e.fb1 = F(pi++);
                          e.fb2 = F(pi++); e.fw1 = F(pi++); e.fw2 = F(pi++);
                          e.ln1b = F(pi++); e.ln1g = F(pi++); e.ln2b = F(pi++);
                          e.ln2g = F(pi++); e.wo = F(pi++); e.wqkv = F(pi++); return e; };
    auto getSmb = [&]() { SmbP s; s.fa_b1 = F(pi++); s.fa_b2 = F(pi++); s.fa_w1 = F(pi++);
                          s.fa_w2 = F(pi++); s.fb_b1 = F(pi++); s.fb_b2 = F(pi++);
                          s.fb_w1 = F(pi++); s.fb_w2 = F(pi++); s.m_b1 = F(pi++);
                          s.m_b2 = F(pi++); s.m_router = F(pi++); s.m_w1 = F(pi++);
                          s.m_w2 = F(pi++); return s; };

    HeadP content_head = getHead();
    const float* fov_emb = F(pi++);
    HeadP geo_head = getHead();
    EncP GL[4]; for (int i = 0; i < 4; i++) GL[i] = getEnc();
    const float* geocode_emb = F(pi++);
    HeadP head_fov = getHead();
    HeadP head_geo = getHead();
    HeadP head_sem = getHead();
    const float* img_emb = F(pi++);
    HeadP img_head = getHead();
    SmbP img_moe = getSmb();
    EncP ML[4]; for (int i = 0; i < 4; i++) ML[i] = getEnc();
    const float* modality = F(pi++);
    HeadP query_head = getHead();
    SmbP text_moe = getSmb();
    const int* img_idx = (const int*)d_in[pi++];
    const int* fov_idx = (const int*)d_in[pi++];
    const int* geocode = (const int*)d_in[pi++];

    float* out = (float*)d_out;
    const long long OUT_SEM = 0, OUT_GEO = 2048LL * 512,
                    OUT_FOV = 2LL * 2048 * 512, OUT_Q = 2LL * 2048 * 512 + 2048LL * 4;

    // workspace layout (floats)
    float* ws = (float*)d_ws;
    WS w;
    long long off = 0;
    w.X     = ws + off; off += 3145728;    // 2048*1536 == 6144*512
    w.T1    = ws + off; off += 8388608;    // 2048*4096
    w.X2    = ws + off; off += 3145728;
    w.EB    = ws + off; off += 6291456;    // 4096*1536
    w.HB    = ws + off; off += 16777216;   // 4096*4096 (also qkv/ffh for 6144 tokens)
    w.EO    = ws + off; off += 6291456;
    w.VIMG  = ws + off; off += 1048576;
    w.VCONT = ws + off; off += 1048576;
    w.VGEO  = ws + off; off += 1048576;
    w.GPOOL = ws + off; off += 1048576;
    w.MPOOL = ws + off; off += 1048576;
    w.G1  = ws + off; off += 2048;
    w.G2  = ws + off; off += 2048;
    w.W1O = ws + off; off += 2048;
    w.W2O = ws + off; off += 2048;
    w.E1 = (int*)(ws + off); off += 2048;
    w.E2 = (int*)(ws + off); off += 2048;
    w.D1 = (int*)(ws + off); off += 2048;
    w.D2 = (int*)(ws + off); off += 2048;

    const int B = 2048, DEMB = 1536, DIM = 512;

    // ---- img stream ----
    k_img_input<<<(B * DEMB + 255) / 256, 256, 0, stream>>>(
        img_vec, img_emb, fov_emb, img_idx, fov_idx, w.X, B, DEMB);
    run_smb(stream, w.X, img_moe, w);
    run_head(stream, w.X2, img_head, w.VIMG, DEMB, 1024, DIM, B, w.X, w.T1);

    // ---- content stream ----
    run_smb(stream, content_vec, text_moe, w);
    run_head(stream, w.X2, content_head, w.VCONT, DEMB, 1024, DIM, B, w.X, w.T1);

    // ---- query stream (writes d_out directly) ----
    run_smb(stream, query_vec, text_moe, w);
    run_head(stream, w.X2, query_head, out + OUT_Q, DEMB, 1024, DIM, B, w.X, w.T1);

    // ---- geo stream (chunked over batch: 4 x 512) ----
    for (int c = 0; c < 4; c++) {
        const int b0 = c * 512, nb = 512, T = nb * 12;
        k_geo_embed<<<(T * DIM + 255) / 256, 256, 0, stream>>>(geocode, geocode_emb, w.X, b0, nb);
        for (int l = 0; l < 4; l++)
            run_enc(stream, w.X, w.HB, w.X2, w.EB, GL[l], T, nb, 12);
        k_meanpool<<<(nb * DIM + 255) / 256, 256, 0, stream>>>(
            w.X, w.GPOOL + (long long)b0 * DIM, nb, 12, DIM);
    }
    run_head(stream, w.GPOOL, geo_head, w.VGEO, DIM, 1024, DIM, B, w.X, w.T1);

    // ---- merge stream ----
    k_build_merge<<<(B * 3 * DIM + 255) / 256, 256, 0, stream>>>(
        w.VIMG, w.VCONT, w.VGEO, modality, w.X, B, DIM);
    for (int l = 0; l < 4; l++)
        run_enc(stream, w.X, w.HB, w.X2, w.EB, ML[l], B * 3, B, 3);
    k_meanpool<<<(B * DIM + 255) / 256, 256, 0, stream>>>(w.X, w.MPOOL, B, 3, DIM);

    run_head(stream, w.MPOOL, head_sem, out + OUT_SEM, DIM, 1024, DIM, B, w.X, w.T1);
    run_head(stream, w.MPOOL, head_geo, out + OUT_GEO, DIM, 1024, DIM, B, w.X, w.T1);
    run_head(stream, w.MPOOL, head_fov, out + OUT_FOV, DIM, 1024, 4, B, w.X, w.T1);
}
